// GAE_11261404250405
// MI455X (gfx1250) — matrix operand order repaired and checked by reading, not compiled, not run on hardware
//
#include <hip/hip_runtime.h>

typedef __attribute__((ext_vector_type(16))) _Float16 v16h;
typedef __attribute__((ext_vector_type(8)))  float    v8f;

#define N_NODES_C 6000
#define NUM_USER_C 2000
#define N_ITEM_C 4000
#define HID_C 64
#define OUT_C 64
#define N_REL_C 5
#define N_BASIS_C 2
#define N_EDGES_C 1000000

union Frag { v16h h; unsigned int u[8]; };

// ---- CDNA5 async global->LDS tile staging (ASYNCcnt-tracked, bypasses VGPRs) ----
// Inline asm sidesteps the builtin's address-space-typed parameters.
// VFLAT form: global_load_async_to_lds_b128 vDst(LDS byte addr), v[a:a+1](global addr), off
__device__ __forceinline__ void async_ld16(const void* gp, void* lp) {
    unsigned int lds      = (unsigned int)(unsigned long long)lp;  // low 32 bits = LDS offset
    unsigned long long ga = (unsigned long long)gp;
    asm volatile("global_load_async_to_lds_b128 %0, %1, off"
                 :: "v"(lds), "v"(ga)
                 : "memory");
}
__device__ __forceinline__ void wait_async0() {
#if defined(__has_builtin)
#if __has_builtin(__builtin_amdgcn_s_wait_asynccnt)
    __builtin_amdgcn_s_wait_asynccnt(0);
#else
    asm volatile("s_wait_asynccnt 0x0" ::: "memory");
#endif
#else
    asm volatile("s_wait_asynccnt 0x0" ::: "memory");
#endif
}

// ---------------- Phase 1: cumsum over relations -> weight table ----------------
__global__ void k_cumsum(const float* __restrict__ ordb, float* __restrict__ w) {
    int t = blockIdx.x * 256 + threadIdx.x;
    if (t >= N_NODES_C * HID_C) return;
    float acc = 0.f;
#pragma unroll
    for (int r = 0; r < N_REL_C; ++r) {
        acc += ordb[(long)r * N_NODES_C * HID_C + t];
        w[(long)r * N_NODES_C * HID_C + t] = acc;
    }
}

__global__ void k_zero(float* __restrict__ p, int n) {
    int t = blockIdx.x * 256 + threadIdx.x;
    if (t < n) p[t] = 0.f;
}

// ---------------- Phase 2: gather/scale/scatter-add over edges ----------------
// 32 lanes per edge, each lane does columns lane and lane+32.
__global__ void k_scatter(const int* __restrict__ eidx, const int* __restrict__ etype,
                          const float* __restrict__ enorm, const float* __restrict__ w,
                          float* __restrict__ feats) {
    unsigned g = blockIdx.x * 256u + threadIdx.x;
    unsigned e = g >> 5;
    unsigned lane = g & 31u;
    if (e >= N_EDGES_C) return;
    // pull edge records 2048 edges (256 blocks) ahead toward L2 (global_prefetch_b8)
    if (lane == 0 && e + 2048 < N_EDGES_C) {
        __builtin_prefetch(&eidx[e + 2048], 0, 1);
        __builtin_prefetch(&eidx[N_EDGES_C + e + 2048], 0, 1);
        __builtin_prefetch(&enorm[e + 2048], 0, 1);
    }
    int src = eidx[e];
    int dst = eidx[N_EDGES_C + e];
    int et  = etype[e];
    float nm = enorm[e];
    const float* wr = w + ((long)et * N_NODES_C + src) * HID_C;
    float* fr = feats + (long)dst * HID_C;
    atomicAdd(&fr[lane],      wr[lane]      * nm);
    atomicAdd(&fr[lane + 32], wr[lane + 32] * nm);
}

// ---------------- Phase 3: g = relu(relu(feats) @ W^T), all 6000 rows ----------------
__global__ void k_fc(const float* __restrict__ feats, const float* __restrict__ W,
                     float* __restrict__ g) {
    __shared__ float fr[HID_C];
    int node = blockIdx.x;
    int d = threadIdx.x;
    float v = feats[(long)node * HID_C + d];
    fr[d] = v > 0.f ? v : 0.f;
    __syncthreads();
    float acc = 0.f;
#pragma unroll
    for (int h = 0; h < HID_C; ++h) acc += fr[h] * W[d * HID_C + h];
    g[(long)node * HID_C + d] = acc > 0.f ? acc : 0.f;
}

// ---------------- Phase 4a: Q = coefs @ basis_matrix ----------------
__global__ void k_q(const float* __restrict__ coefs, const float* __restrict__ bas,
                    float* __restrict__ Q) {
    int t = blockIdx.x * 256 + threadIdx.x;
    if (t >= N_REL_C * OUT_C * OUT_C) return;
    int r = t / (OUT_C * OUT_C);
    int j = t % (OUT_C * OUT_C);
    Q[t] = coefs[r * N_BASIS_C + 0] * bas[j] + coefs[r * N_BASIS_C + 1] * bas[OUT_C * OUT_C + j];
}

// ---------------- Phase 4b: Ur[r][u][e] = sum_d g[u][d] * Q[r][d][e]  (f16 out) ----------------
__global__ void k_ur(const float* __restrict__ g, const float* __restrict__ Q,
                     _Float16* __restrict__ urh) {
    __shared__ float uu[HID_C];
    int u = blockIdx.x;
    int r = blockIdx.y;
    int e = threadIdx.x;
    uu[e] = g[(long)u * HID_C + e];
    __syncthreads();
    float acc = 0.f;
#pragma unroll
    for (int d = 0; d < HID_C; ++d) acc += uu[d] * Q[((long)r * OUT_C + d) * OUT_C + e];
    urh[((long)r * NUM_USER_C + u) * OUT_C + e] = (_Float16)acc;
}

// ---------------- Phase 4c: item features -> f16 ----------------
__global__ void k_ih(const float* __restrict__ g, _Float16* __restrict__ ihh) {
    int t = blockIdx.x * 256 + threadIdx.x;
    if (t < N_ITEM_C * HID_C) ihh[t] = (_Float16)g[(long)NUM_USER_C * HID_C + t];
}

// ---------------- Phase 5: out[u,i,r] = Ur[r,u,:] . ih[i,:]  via WMMA ----------------
// Block = 256 threads = 8 waves laid out 2(u) x 4(i); each wave owns a 16x16 tile
// with 5 accumulators (one per relation). K = 64 -> two 16x16x32 f16 WMMA steps.
__global__ __launch_bounds__(256) void k_einsum(const unsigned int* __restrict__ urh32,
                                                const unsigned int* __restrict__ ihh32,
                                                float* __restrict__ out) {
    // shA: [r][um(32)][kp(32)] f16 pairs ; shB: [in(64)][kp(32)] f16 pairs
    __shared__ __align__(16) unsigned int shA[N_REL_C * 32 * 32];
    __shared__ __align__(16) unsigned int shB[64 * 32];

    const int i0g = blockIdx.x * 64;
    const int u0g = blockIdx.y * 32;

    // Async tile staging: 16B chunks, global -> LDS directly (tracked by ASYNCcnt).
    // shA = 1280 chunks, shB = 512 chunks; 256 threads.
    for (int c = threadIdx.x; c < N_REL_C * 32 * 8; c += 256) {
        int r   = c >> 8;          // 32*8 = 256 chunks per relation
        int rem = c & 255;
        int um  = rem >> 3;
        int kq  = rem & 7;
        int urow = u0g + um;
        if (urow > NUM_USER_C - 1) urow = NUM_USER_C - 1;   // clamp; guarded at store
        const unsigned int* gp = urh32 + ((long)r * NUM_USER_C + urow) * 32 + kq * 4;
        async_ld16(gp, &shA[c * 4]);
    }
    for (int c = threadIdx.x; c < 64 * 8; c += 256) {
        int in_ = c >> 3;
        int kq  = c & 7;
        int irow = i0g + in_;
        if (irow > N_ITEM_C - 1) irow = N_ITEM_C - 1;        // clamp; guarded at store
        const unsigned int* gp = ihh32 + (long)irow * 32 + kq * 4;
        async_ld16(gp, &shB[c * 4]);
    }
    wait_async0();
    __syncthreads();

    const int wave = threadIdx.x >> 5;
    const int lane = threadIdx.x & 31;
    const int wu = wave >> 2;     // 0..1  (u sub-tile)
    const int wi = wave & 3;      // 0..3  (i sub-tile)
    const int hi = (lane >> 4) & 1;
    const int mn = lane & 15;     // M for A, N for B/C/D

    // B fragments (contiguous K, same as A: lanes 0-15 hold K=0..15, lanes 16-31 K=16..31; N = lane&15)
    Frag b0, b1;
    const int rowB = (wi * 16 + mn) * 32;
#pragma unroll
    for (int v = 0; v < 8; ++v) {
        b0.u[v] = shB[rowB + (hi ? 8 : 0) + v];          // k-step 0: K 0..31
        b1.u[v] = shB[rowB + 16 + (hi ? 8 : 0) + v];     // k-step 1: K 32..63
    }

    v8f acc[N_REL_C] = {};

#pragma unroll
    for (int r = 0; r < N_REL_C; ++r) {
        // A fragments (M = lane&15; lanes 0-15 hold K=0..15, lanes 16-31 K=16..31;
        // pairs of consecutive K per vgpr)
        Frag a0, a1;
        const int rowA = (r * 32 + wu * 16 + mn) * 32;
#pragma unroll
        for (int v = 0; v < 8; ++v) {
            int kp0 = (hi ? 8 : 0) + v;
            a0.u[v] = shA[rowA + kp0];
            a1.u[v] = shA[rowA + 16 + kp0];
        }
        acc[r] = __builtin_amdgcn_wmma_f32_16x16x32_f16(false, a0.h, false, b0.h,
                                                        (short)0, acc[r], false, false);
        acc[r] = __builtin_amdgcn_wmma_f32_16x16x32_f16(false, a1.h, false, b1.h,
                                                        (short)0, acc[r], false, false);
    }

    // Store: lane holds (m = v + 8*hi, n = lane&15) for each accumulator.
    // Output layout (u*4000 + i)*5 + r -> 5 consecutive floats per (u,i).
    const int col = i0g + wi * 16 + mn;
#pragma unroll
    for (int v = 0; v < 8; ++v) {
        int row = u0g + wu * 16 + v + (hi ? 8 : 0);
        if (row < NUM_USER_C && col < N_ITEM_C) {
            float* p = out + ((long)row * N_ITEM_C + col) * N_REL_C;
#pragma unroll
            for (int r = 0; r < N_REL_C; ++r)
                __builtin_nontemporal_store(acc[r][v], p + r);  // NT: don't pollute L2
        }
    }
}

extern "C" void kernel_launch(void* const* d_in, const int* in_sizes, int n_in,
                              void* d_out, int out_size, void* d_ws, size_t ws_size,
                              hipStream_t stream) {
    // setup_inputs order: x, edge_index, edge_type, edge_norm, ord_basis, fc_w, basis_matrix, coefs
    const int*   eidx  = (const int*)d_in[1];
    const int*   etype = (const int*)d_in[2];
    const float* enorm = (const float*)d_in[3];
    const float* ordb  = (const float*)d_in[4];
    const float* fcw   = (const float*)d_in[5];
    const float* bas   = (const float*)d_in[6];
    const float* coefs = (const float*)d_in[7];
    float* out = (float*)d_out;

    char* ws = (char*)d_ws;
    size_t off = 0;
    auto carve = [&](size_t bytes) -> void* {
        void* p = ws + off;
        off = (off + bytes + 255) & ~(size_t)255;
        return p;
    };
    float*    w     = (float*)carve(sizeof(float) * N_REL_C * N_NODES_C * HID_C);   // 7.68 MB
    float*    feats = (float*)carve(sizeof(float) * N_NODES_C * HID_C);             // 1.54 MB
    float*    g     = (float*)carve(sizeof(float) * N_NODES_C * HID_C);             // 1.54 MB
    float*    Q     = (float*)carve(sizeof(float) * N_REL_C * OUT_C * OUT_C);       // 80 KB
    _Float16* urh   = (_Float16*)carve(sizeof(_Float16) * N_REL_C * NUM_USER_C * OUT_C); // 1.28 MB
    _Float16* ihh   = (_Float16*)carve(sizeof(_Float16) * N_ITEM_C * OUT_C);        // 0.51 MB

    k_cumsum<<<(N_NODES_C * HID_C + 255) / 256, 256, 0, stream>>>(ordb, w);
    k_zero<<<(N_NODES_C * HID_C + 255) / 256, 256, 0, stream>>>(feats, N_NODES_C * HID_C);

    long scat_threads = (long)N_EDGES_C * 32;
    k_scatter<<<(unsigned)((scat_threads + 255) / 256), 256, 0, stream>>>(eidx, etype, enorm, w, feats);

    k_fc<<<N_NODES_C, HID_C, 0, stream>>>(feats, fcw, g);
    k_q<<<(N_REL_C * OUT_C * OUT_C + 255) / 256, 256, 0, stream>>>(coefs, bas, Q);

    dim3 gur(NUM_USER_C, N_REL_C);
    k_ur<<<gur, OUT_C, 0, stream>>>(g, Q, urh);
    k_ih<<<(N_ITEM_C * HID_C + 255) / 256, 256, 0, stream>>>(g, ihh);

    dim3 ge((N_ITEM_C + 63) / 64, (NUM_USER_C + 31) / 32);
    k_einsum<<<ge, 256, 0, stream>>>((const unsigned int*)urh, (const unsigned int*)ihh, out);
}
